// EncoderLayer_45973329936959
// MI455X (gfx1250) — matrix operand order repaired and checked by reading, not compiled, not run on hardware
//
#include <hip/hip_runtime.h>

// Problem constants (reference: N,L,H,E,A,F = 2,2048,1024,64,16,4096)
#define NB 2
#define LS 2048
#define HD 1024
#define ED 64
#define AH 16
#define FD 4096
#define MROWS (NB * LS)          // 4096
#define LN_EPS 1e-5f
#define ATTN_SCALE 0.022097086912079608f  // 1/sqrt(2048) (reference uses 1/sqrt(L))

#ifndef __has_builtin
#define __has_builtin(x) 0
#endif

typedef __attribute__((ext_vector_type(16))) __bf16 v16bf;
typedef __attribute__((ext_vector_type(8)))  __bf16 v8bf;
typedef __attribute__((ext_vector_type(8)))  float  v8f;

// ---- WMMA helpers -----------------------------------------------------------
// A-fragment (16x32 bf16): lane<16 holds row m=lane, k = 0..15;
// lane>=16 holds k = 16..31.  p points at [row][base]; 16 contiguous k.
static __device__ __forceinline__ v16bf frag_a(const __bf16* p) {
  union { v16bf v; v8bf h[2]; } u;
  u.h[0] = *(const v8bf*)(p);
  u.h[1] = *(const v8bf*)(p + 8);
  return u.v;
}
// B-fragment (32x16 bf16) from a [n][k] (k-contiguous) buffer: lane col n=lane&15,
// k = (lane>>4)*16 .. +15 contiguous.
static __device__ __forceinline__ v16bf frag_b(const __bf16* p) {
  union { v16bf v; v8bf h[2]; } u;
  u.h[0] = ((const v8bf*)p)[0];
  u.h[1] = ((const v8bf*)p)[1];
  return u.v;
}
#define WMMA_BF16(C, Af, Bf) \
  __builtin_amdgcn_wmma_f32_16x16x32_bf16(false, (Af), false, (Bf), (short)0, (C), false, false)

// ---- Tensor Data Mover (async tile DMA -> LDS) ------------------------------
#if __has_builtin(__builtin_amdgcn_tensor_load_to_lds)
#define USE_TDM 1
#else
#define USE_TDM 0
#endif

#if __has_builtin(__builtin_amdgcn_s_wait_tensorcnt)
#define WAIT_TENSOR(n) __builtin_amdgcn_s_wait_tensorcnt(n)
#else
#define WAIT_TENSOR(n) asm volatile("s_wait_tensorcnt %0" ::"i"(n) : "memory")
#endif

#if USE_TDM
typedef __attribute__((ext_vector_type(4))) unsigned int u32x4;
typedef __attribute__((ext_vector_type(4))) int i32x4;
typedef __attribute__((ext_vector_type(8))) int i32x8;

// 2D bf16 tile DMA: rows x rowlen elements, row pitch = stride elements.
// pad_int_code / pad_amt_code follow D# group1 encodings (ISA 08 §8.4).
static __device__ __forceinline__ void tdm_load_2d(const __bf16* gsrc, unsigned lds_addr,
                                                   int rows, int rowlen, int stride,
                                                   int pad_int_code, int pad_amt_code) {
  unsigned long long ga = (unsigned long long)(const void*)gsrc;
  u32x4 g0;
  g0[0] = 1u;                                  // count=1, user descriptor
  g0[1] = lds_addr;                            // LDS byte address
  g0[2] = (unsigned)ga;                        // global_addr[31:0]
  g0[3] = (unsigned)(ga >> 32) | (2u << 30);   // global_addr[56:32] | type=2 ("image")
  unsigned td0 = (unsigned)stride;             // tensor_dim0 (covers tile)
  unsigned td1 = 1u << 20;                     // tensor_dim1 (large, always in-bounds)
  i32x8 g1;
  g1[0] = (int)((1u << 16) | (1u << 20) |      // data_size=2B | pad_enable
                ((unsigned)pad_int_code << 22) | ((unsigned)pad_amt_code << 25));
  g1[1] = (int)((td0 & 0xFFFFu) << 16);                      // tensor_dim0 lo
  g1[2] = (int)((td0 >> 16) | ((td1 & 0xFFFFu) << 16));      // dim0 hi | dim1 lo
  g1[3] = (int)((td1 >> 16) | ((unsigned)rowlen << 16));     // dim1 hi | tile_dim0
  g1[4] = (int)(unsigned)rows;                               // tile_dim1 (tile_dim2=0)
  g1[5] = (int)(unsigned)stride;                             // tensor_dim0_stride lo
  g1[6] = 0;
  g1[7] = 0;
  i32x4 gz = {0, 0, 0, 0};
#if __clang_major__ >= 23
  i32x8 gz8 = {0, 0, 0, 0, 0, 0, 0, 0};
  __builtin_amdgcn_tensor_load_to_lds(g0, g1, gz, gz, gz8, 0);
#else
  __builtin_amdgcn_tensor_load_to_lds(g0, g1, gz, gz, 0);
#endif
}
#endif  // USE_TDM

// ---- conversion / layout kernels -------------------------------------------
__global__ __launch_bounds__(256) void cvt_bf16_kernel(const float* __restrict__ src,
                                                       __bf16* __restrict__ dst, int n) {
  int i = (blockIdx.x * 256 + threadIdx.x) * 4;
  if (i + 3 < n) {
    float4 f = *(const float4*)(src + i);
    dst[i + 0] = (__bf16)f.x; dst[i + 1] = (__bf16)f.y;
    dst[i + 2] = (__bf16)f.z; dst[i + 3] = (__bf16)f.w;
  }
}

// Wq/Wk/Wv: (A,H,E) f32 -> bf16 [n = a*E+e][k = h]
__global__ __launch_bounds__(256) void permute_qkvw_kernel(const float* __restrict__ src,
                                                           __bf16* __restrict__ dst) {
  int idx = blockIdx.x * 256 + threadIdx.x;       // over A*H*E = 1M
  int a = idx >> 16;
  int r = idx & 65535;
  int h = r >> 6, e = r & 63;
  dst[(size_t)((a << 6) + e) * HD + h] = (__bf16)src[idx];
}

// generic f32 [K][Nn] -> bf16 [Nn][K]
__global__ __launch_bounds__(256) void transpose_cvt_kernel(const float* __restrict__ src,
                                                            __bf16* __restrict__ dst,
                                                            int K, int Nn) {
  int idx = blockIdx.x * 256 + threadIdx.x;
  if (idx < K * Nn) {
    int k = idx / Nn, n = idx % Nn;
    dst[(size_t)n * K + k] = (__bf16)src[idx];
  }
}

// ---- GEMM: C[M][Nn] = A[M][K] * B^T   (B stored [Nn][K]) -------------------
// 256 threads = 8 waves; block tile 128x128, K step 32; wave tile 32x64 (2x4 WMMA)
// TDM path: wave0 DMAs the next K-stage tiles while all waves run WMMAs (double buffer).
template <int MODE>
__global__ __launch_bounds__(256)
void gemm_kernel(const __bf16* __restrict__ Amat, const __bf16* __restrict__ Bmat,
                 const float* __restrict__ bias, const float* __restrict__ resid,
                 float* __restrict__ outF, __bf16* __restrict__ outB, int Nn, int K) {
  __shared__ __bf16 As[2][128][40];
  __shared__ __bf16 Bs[2][128][40];

  const int tid  = threadIdx.x;
  const int lane = tid & 31;
  const int wid  = tid >> 5;
  const int wm   = wid & 3;
  const int wn   = wid >> 2;
  const int half = lane >> 4;
  const int l15  = lane & 15;
  const int bm   = blockIdx.x * 128;
  const int bn   = blockIdx.y * 128;

  v8f acc[2][4] = {};

#if USE_TDM
  const __bf16* gA = Amat + (size_t)bm * K;
  const __bf16* gB = Bmat + (size_t)bn * K;
  if (wid == 0) {  // prologue: stage 0 into buffer 0
    tdm_load_2d(gA, (unsigned)(size_t)&As[0][0][0], 128, 32, K, 3, 3);
    tdm_load_2d(gB, (unsigned)(size_t)&Bs[0][0][0], 128, 32, K, 3, 3);
  }
  for (int k0 = 0, it = 0; k0 < K; k0 += 32, ++it) {
    const int buf = it & 1;
    __syncthreads();                       // all readers of buf^1 finished
    if (wid == 0) {
      if (k0 + 32 < K) {
        tdm_load_2d(gA + k0 + 32, (unsigned)(size_t)&As[buf ^ 1][0][0], 128, 32, K, 3, 3);
        tdm_load_2d(gB + k0 + 32, (unsigned)(size_t)&Bs[buf ^ 1][0][0], 128, 32, K, 3, 3);
        WAIT_TENSOR(2);                    // current stage landed; next 2 in flight
      } else {
        WAIT_TENSOR(0);
      }
    }
    __syncthreads();                       // current buf ready for everyone
#else
  for (int k0 = 0; k0 < K; k0 += 32) {
    const int buf = 0;
    const int ldr = tid >> 1;
    const int ldk = (tid & 1) * 16;
    const __bf16* ga = Amat + (size_t)(bm + ldr) * K + k0 + ldk;
    const __bf16* gb = Bmat + (size_t)(bn + ldr) * K + k0 + ldk;
    ((v8bf*)&As[0][ldr][ldk])[0] = ((const v8bf*)ga)[0];
    ((v8bf*)&As[0][ldr][ldk])[1] = ((const v8bf*)ga)[1];
    ((v8bf*)&Bs[0][ldr][ldk])[0] = ((const v8bf*)gb)[0];
    ((v8bf*)&Bs[0][ldr][ldk])[1] = ((const v8bf*)gb)[1];
    if (k0 + 32 < K) {
      __builtin_prefetch(ga + 32, 0, 0);
      __builtin_prefetch(gb + 32, 0, 0);
    }
    __syncthreads();
#endif

    v16bf af[2], bf_[4];
#pragma unroll
    for (int mt = 0; mt < 2; ++mt)
      af[mt] = frag_a(&As[buf][wm * 32 + mt * 16 + l15][half * 16]);
#pragma unroll
    for (int nt = 0; nt < 4; ++nt)
      bf_[nt] = frag_b(&Bs[buf][wn * 64 + nt * 16 + l15][half * 16]);
#pragma unroll
    for (int mt = 0; mt < 2; ++mt)
#pragma unroll
      for (int nt = 0; nt < 4; ++nt)
        acc[mt][nt] = WMMA_BF16(acc[mt][nt], af[mt], bf_[nt]);
#if !USE_TDM
    __syncthreads();
#endif
  }

  // epilogue: C layout -> row = (lane>>4)*8 + i, col = lane&15
#pragma unroll
  for (int mt = 0; mt < 2; ++mt) {
#pragma unroll
    for (int nt = 0; nt < 4; ++nt) {
#pragma unroll
      for (int i = 0; i < 8; ++i) {
        int row = bm + wm * 32 + mt * 16 + half * 8 + i;
        int col = bn + wn * 64 + nt * 16 + l15;
        float v = acc[mt][nt][i] + bias[col];
        if constexpr (MODE == 0) {
          int n = row >> 11, l = row & 2047;      // L = 2048
          int a = col >> 6,  e = col & 63;        // E = 64
          outB[(((size_t)n * AH + a) * LS + l) * ED + e] = (__bf16)v;
        } else if constexpr (MODE == 1) {
          v += resid[(size_t)row * Nn + col];
          outF[(size_t)row * Nn + col] = v;
          outB[(size_t)row * Nn + col] = (__bf16)v;
        } else if constexpr (MODE == 2) {
          v = fmaxf(v, 0.0f);
          outB[(size_t)row * Nn + col] = (__bf16)v;
        } else {
          outF[(size_t)row * Nn + col] = v;
        }
      }
    }
  }
}

// ---- flash attention --------------------------------------------------------
// grid (L/64, A, N); 128 threads = 4 waves, one 16-row Q tile per wave.
// 64-key blocks; K tile double-buffered via TDM; row-sums come free from an
// all-ones B tile in the P*V WMMAs (no cross-lane sum reduction needed).
__global__ __launch_bounds__(128)
void flash_attn_kernel(const __bf16* __restrict__ q, const __bf16* __restrict__ k,
                       const __bf16* __restrict__ v, const long long* __restrict__ mask,
                       __bf16* __restrict__ ctx) {
  __shared__ __bf16 Ks[2][64][72];     // [key][e]   (144B row pitch, DMA-padded)
  __shared__ __bf16 Vs[64][72];        // [e][key]   (transposed by hand)
  __shared__ __bf16 Ps[4][16][72];     // per-wave P [qrow][key]

  const int tid  = threadIdx.x;
  const int lane = tid & 31;
  const int wid  = tid >> 5;
  const int half = lane >> 4;
  const int l15  = lane & 15;
  const int n    = blockIdx.z;
  const int a    = blockIdx.y;
  const int q0   = blockIdx.x * 64 + wid * 16;
  const size_t head = (size_t)n * AH + a;

  // Q fragments for this wave's 16 rows, K-dim = e (two 32-wide steps)
  const __bf16* gq = q + (head * LS + q0 + l15) * ED;
  v16bf aq[2];
#pragma unroll
  for (int es = 0; es < 2; ++es)
    aq[es] = frag_a(gq + es * 32 + half * 16);

  v16bf ones;
#pragma unroll
  for (int j = 0; j < 16; ++j) ones[j] = (__bf16)1.0f;

  bool rowvalid[8];
  float m_run[8], alpha[8];
#pragma unroll
  for (int i = 0; i < 8; ++i) {
    rowvalid[i] = (mask[(size_t)n * LS + q0 + half * 8 + i] != 0);
    m_run[i] = -1e30f;
  }
  v8f acc[4] = {};
  v8f acc_l = {};

  const int krow = tid >> 1;           // 0..63
  const int ke0  = (tid & 1) * 32;     // 0 / 32

#if USE_TDM
  if (wid == 0)
    tdm_load_2d(k + head * LS * ED, (unsigned)(size_t)&Ks[0][0][0], 64, 64, 64, 4, 3);
#endif

  for (int kb = 0; kb < LS; kb += 64) {
    const int buf = (kb >> 6) & 1;
    __syncthreads();                   // everyone done with Ks[buf^1] and Vs
#if USE_TDM
    if (wid == 0 && kb + 64 < LS)
      tdm_load_2d(k + (head * LS + kb + 64) * ED, (unsigned)(size_t)&Ks[buf ^ 1][0][0],
                  64, 64, 64, 4, 3);
#else
    {
      const __bf16* gk = k + (head * LS + kb + krow) * ED + ke0;
#pragma unroll
      for (int c = 0; c < 4; ++c)
        ((v8bf*)&Ks[buf][krow][ke0])[c] = ((const v8bf*)gk)[c];
    }
#endif
    // V tile, transposed to [e][key] by hand
    {
      const __bf16* gv = v + (head * LS + kb + krow) * ED + ke0;
      v8bf va[4];
#pragma unroll
      for (int c = 0; c < 4; ++c) va[c] = ((const v8bf*)gv)[c];
#pragma unroll
      for (int c = 0; c < 4; ++c)
#pragma unroll
        for (int j = 0; j < 8; ++j) Vs[ke0 + c * 8 + j][krow] = va[c][j];
    }
#if USE_TDM
    if (wid == 0) {
      if (kb + 64 < LS) WAIT_TENSOR(1); else WAIT_TENSOR(0);
    }
#endif
    __syncthreads();                   // K tile (buf) + V tile ready

    // S = Q * K^T : 4 key sub-tiles x 2 e-steps
    v8f s[4] = {};
#pragma unroll
    for (int es = 0; es < 2; ++es) {
#pragma unroll
      for (int j = 0; j < 4; ++j) {
        v16bf b = frag_b(&Ks[buf][j * 16 + l15][es * 32 + half * 16]);
        s[j] = WMMA_BF16(s[j], aq[es], b);
      }
    }

    // online softmax: only the row-max needs a cross-lane reduction
#pragma unroll
    for (int i = 0; i < 8; ++i) {
      float xs[4];
#pragma unroll
      for (int j = 0; j < 4; ++j) {
        xs[j] = s[j][i] * ATTN_SCALE;
        if (!rowvalid[i]) xs[j] = -1e30f;      // query-axis mask quirk
      }
      float mt = fmaxf(fmaxf(xs[0], xs[1]), fmaxf(xs[2], xs[3]));
#pragma unroll
      for (int off = 1; off < 16; off <<= 1) mt = fmaxf(mt, __shfl_xor(mt, off, 32));
      float mn = fmaxf(m_run[i], mt);
      alpha[i] = __expf(m_run[i] - mn);
      m_run[i] = mn;
#pragma unroll
      for (int j = 0; j < 4; ++j)
        Ps[wid][half * 8 + i][j * 16 + l15] = (__bf16)__expf(xs[j] - mn);
    }
    asm volatile("s_wait_dscnt 0" ::: "memory");  // same-wave LDS RAW before frag read

    v16bf pa0 = frag_a(&Ps[wid][l15][half * 16]);        // keys 0..31
    v16bf pa1 = frag_a(&Ps[wid][l15][32 + half * 16]);   // keys 32..63

    // rescale running stats, then O += P*V and l += rowsum(P) (ones column trick)
#pragma unroll
    for (int i = 0; i < 8; ++i) acc_l[i] *= alpha[i];
    acc_l = WMMA_BF16(acc_l, pa0, ones);
    acc_l = WMMA_BF16(acc_l, pa1, ones);
#pragma unroll
    for (int t = 0; t < 4; ++t) {
#pragma unroll
      for (int i = 0; i < 8; ++i) acc[t][i] *= alpha[i];
      acc[t] = WMMA_BF16(acc[t], pa0, frag_b(&Vs[t * 16 + l15][half * 16]));
      acc[t] = WMMA_BF16(acc[t], pa1, frag_b(&Vs[t * 16 + l15][32 + half * 16]));
    }
  }

  // normalize and write ctx (n, l, a*E+e)
#pragma unroll
  for (int t = 0; t < 4; ++t) {
#pragma unroll
    for (int i = 0; i < 8; ++i) {
      int row = q0 + half * 8 + i;
      float o = acc[t][i] / acc_l[i];
      ctx[((size_t)n * LS + row) * (AH * ED) + a * ED + t * 16 + l15] = (__bf16)o;
    }
  }
}

// ---- LayerNorm(ffn) + attn residual ----------------------------------------
__global__ __launch_bounds__(256)
void ln_residual_kernel(const float* __restrict__ ffn2, const float* __restrict__ attnx,
                        const float* __restrict__ gamma, const float* __restrict__ beta,
                        float* __restrict__ out) {
  __shared__ float red[256];
  const int row = blockIdx.x, tid = threadIdx.x;
  const float* xr = ffn2 + (size_t)row * HD;
  float v[4]; float s = 0.f;
#pragma unroll
  for (int j = 0; j < 4; ++j) { v[j] = xr[tid * 4 + j]; s += v[j]; }
  red[tid] = s; __syncthreads();
  for (int st = 128; st > 0; st >>= 1) { if (tid < st) red[tid] += red[tid + st]; __syncthreads(); }
  float mu = red[0] * (1.0f / HD);
  __syncthreads();
  float s2 = 0.f;
#pragma unroll
  for (int j = 0; j < 4; ++j) { float d = v[j] - mu; s2 += d * d; }
  red[tid] = s2; __syncthreads();
  for (int st = 128; st > 0; st >>= 1) { if (tid < st) red[tid] += red[tid + st]; __syncthreads(); }
  float rstd = rsqrtf(red[0] * (1.0f / HD) + LN_EPS);
#pragma unroll
  for (int j = 0; j < 4; ++j) {
    int c = tid * 4 + j;
    out[(size_t)row * HD + c] =
        (v[j] - mu) * rstd * gamma[c] + beta[c] + attnx[(size_t)row * HD + c];
  }
}

// ---- host -------------------------------------------------------------------
extern "C" void kernel_launch(void* const* d_in, const int* in_sizes, int n_in,
                              void* d_out, int out_size, void* d_ws, size_t ws_size,
                              hipStream_t stream) {
  const float*     x    = (const float*)d_in[0];
  const long long* mask = (const long long*)d_in[1];
  const float* Wq = (const float*)d_in[2];
  const float* bq = (const float*)d_in[3];
  const float* Wk = (const float*)d_in[4];
  const float* bk = (const float*)d_in[5];
  const float* Wv = (const float*)d_in[6];
  const float* bv = (const float*)d_in[7];
  const float* Wo = (const float*)d_in[8];
  const float* bo = (const float*)d_in[9];
  const float* W1 = (const float*)d_in[10];
  const float* b1 = (const float*)d_in[11];
  const float* W2 = (const float*)d_in[12];
  const float* b2 = (const float*)d_in[13];
  const float* gamma = (const float*)d_in[14];
  const float* beta  = (const float*)d_in[15];
  float* out = (float*)d_out;

  char* ws = (char*)d_ws;
  size_t off = 0;
  auto alloc = [&](size_t bytes) -> void* {
    void* p = ws + off;
    off += (bytes + 255) & ~(size_t)255;
    return p;
  };
  const size_t SZ_X  = (size_t)MROWS * HD;   // 4M elems
  const size_t SZ_W  = (size_t)HD * HD;      // 1M
  const size_t SZ_W1 = (size_t)HD * FD;      // 4M
  const size_t SZ_HID= (size_t)MROWS * FD;   // 16M

  __bf16* xb   = (__bf16*)alloc(SZ_X * 2);
  __bf16* wqb  = (__bf16*)alloc(SZ_W * 2);
  __bf16* wkb  = (__bf16*)alloc(SZ_W * 2);
  __bf16* wvb  = (__bf16*)alloc(SZ_W * 2);
  __bf16* wob  = (__bf16*)alloc(SZ_W * 2);
  __bf16* w1b  = (__bf16*)alloc(SZ_W1 * 2);
  __bf16* w2b  = (__bf16*)alloc(SZ_W1 * 2);
  __bf16* qb   = (__bf16*)alloc(SZ_X * 2);
  __bf16* kb   = (__bf16*)alloc(SZ_X * 2);
  __bf16* vb   = (__bf16*)alloc(SZ_X * 2);
  __bf16* ctxb = (__bf16*)alloc(SZ_X * 2);
  float*  attnx= (float*) alloc(SZ_X * 4);
  __bf16* xab  = (__bf16*)alloc(SZ_X * 2);
  __bf16* hid  = (__bf16*)alloc(SZ_HID * 2);
  float*  ffn2 = (float*) alloc(SZ_X * 4);
  (void)ws_size; (void)n_in; (void)in_sizes; (void)out_size;

  // 1) precision conversion + weight layout (all GEMM B matrices become [N][K])
  cvt_bf16_kernel<<<SZ_X / 1024, 256, 0, stream>>>(x, xb, (int)SZ_X);
  permute_qkvw_kernel<<<SZ_W / 256, 256, 0, stream>>>(Wq, wqb);
  permute_qkvw_kernel<<<SZ_W / 256, 256, 0, stream>>>(Wk, wkb);
  permute_qkvw_kernel<<<SZ_W / 256, 256, 0, stream>>>(Wv, wvb);
  transpose_cvt_kernel<<<SZ_W / 256, 256, 0, stream>>>(Wo, wob, HD, HD);
  transpose_cvt_kernel<<<SZ_W1 / 256, 256, 0, stream>>>(W1, w1b, HD, FD);
  transpose_cvt_kernel<<<SZ_W1 / 256, 256, 0, stream>>>(W2, w2b, FD, HD);

  // 2) QKV projections (M=4096, N=1024, K=1024)
  dim3 blk(256);
  dim3 g1024(MROWS / 128, HD / 128);
  gemm_kernel<0><<<g1024, blk, 0, stream>>>(xb, wqb, bq, nullptr, nullptr, qb, HD, HD);
  gemm_kernel<0><<<g1024, blk, 0, stream>>>(xb, wkb, bk, nullptr, nullptr, kb, HD, HD);
  gemm_kernel<0><<<g1024, blk, 0, stream>>>(xb, wvb, bv, nullptr, nullptr, vb, HD, HD);

  // 3) flash attention
  flash_attn_kernel<<<dim3(LS / 64, AH, NB), dim3(128), 0, stream>>>(qb, kb, vb, mask, ctxb);

  // 4) output projection + residual (f32 residual copy + bf16 for FFN)
  gemm_kernel<1><<<g1024, blk, 0, stream>>>(ctxb, wob, bo, x, attnx, xab, HD, HD);

  // 5) FFN
  gemm_kernel<2><<<dim3(MROWS / 128, FD / 128), blk, 0, stream>>>(xab, w1b, b1, nullptr,
                                                                  nullptr, hid, FD, HD);
  gemm_kernel<3><<<g1024, blk, 0, stream>>>(hid, w2b, b2, nullptr, ffn2, nullptr, HD, FD);

  // 6) LayerNorm + residual -> out
  ln_residual_kernel<<<MROWS, 256, 0, stream>>>(ffn2, attnx, gamma, beta, out);
}
